// spnet_33801392620150
// MI455X (gfx1250) — compile-verified, matrix operand order checked by reading, not run on hardware
//
#include <hip/hip_runtime.h>
#include <hip/hip_bf16.h>

// ---------------------------------------------------------------------------
// Problem constants (from the reference):
//   x   : [B=32, C=17, T=48, H=56, W=56] fp32
//   W1  : [1,3,3,17,17]  (DHWIO: [tap][ci][co])
//   W2  : [1,3,3,17,17]
//   Wfc : [60,17], bfc : [60]
//   out : [32,60] fp32
// ---------------------------------------------------------------------------
#define BATCH   32
#define CIN     17
#define TT      48
#define HH      56
#define WW      56
#define THRESH  0.08f

// Implicit-GEMM geometry:
//   K per tap = 32 (17 channels zero-padded), 9 taps
//   N = 48: cols 0..16 -> W1 outputs, cols 17..33 -> W2 outputs, 34..47 zero
#define KPAD    32
#define NPAD    48
#define ROWS_PER_TILE 8                      // 56 = 7 * 8
#define SITES_PER_WG  (ROWS_PER_TILE * WW)   // 448 = 28 M-tiles of 16
#define MTILES        (SITES_PER_WG / 16)

// LDS activation slab: 10 halo rows x 58 halo cols x 24 padded channels
#define LROWS 10
#define LCOLS 58
#define LCH   24
#define XS_ELEMS (LROWS * LCOLS * LCH)       // 13920 ushorts
#define WL_ELEMS (9 * NPAD * KPAD)           // 13824 ushorts
#define SMEM_BYTES ((XS_ELEMS + WL_ELEMS) * 2 + (SITES_PER_WG + NPAD) * 4) // 57472

typedef __attribute__((ext_vector_type(16))) __bf16          v16bf;
typedef __attribute__((ext_vector_type(8)))  float           v8f;
typedef __attribute__((ext_vector_type(8)))  unsigned short  u16x8;

union Frag16 { u16x8 h[2]; v16bf v; };

__device__ __forceinline__ unsigned short f2bf(float f) {
    union { float f; unsigned u; } v; v.f = f;
    unsigned r = v.u + 0x7FFFu + ((v.u >> 16) & 1u);   // round-to-nearest-even
    return (unsigned short)(r >> 16);
}

// B fragment: column `co` of the [K=32 x N=48] tap matrix, bf16 16x16x32 layout.
// wl layout: [tap][co][ci(32)] contiguous in ci -> two 16B LDS loads per lane.
__device__ __forceinline__ v16bf load_b_frag(const unsigned short* wl_s,
                                             int tap, int co, bool hi) {
    const unsigned short* q = wl_s + ((tap * NPAD + co) * KPAD);
    Frag16 B;
    if (!hi) { B.h[0] = *(const u16x8*)(q);     B.h[1] = *(const u16x8*)(q + 16); }
    else     { B.h[0] = *(const u16x8*)(q + 8); B.h[1] = *(const u16x8*)(q + 24); }
    return B.v;
}

// ---------------------------------------------------------------------------
// Kernel 1: zero the per-batch pooled accumulator [32][17] in workspace
// ---------------------------------------------------------------------------
__global__ void spnet_zero_kernel(float* pooledSum) {
    for (int i = threadIdx.x; i < BATCH * CIN; i += blockDim.x)
        pooledSum[i] = 0.0f;
}

// ---------------------------------------------------------------------------
// Kernel 2: pack W1|W2 into bf16 table wl[tap][co(48)][ci(32)]
// ---------------------------------------------------------------------------
__global__ void spnet_wprep_kernel(const float* __restrict__ W1,
                                   const float* __restrict__ W2,
                                   unsigned short* __restrict__ wl) {
    int idx = blockIdx.x * blockDim.x + threadIdx.x;
    if (idx >= 9 * NPAD * KPAD) return;
    int tap = idx / (NPAD * KPAD);
    int rem = idx % (NPAD * KPAD);
    int co  = rem / KPAD;
    int ci  = rem % KPAD;
    float v = 0.0f;
    if (ci < CIN) {
        if (co < CIN)            v = W1[(tap * CIN + ci) * CIN + co];
        else if (co < 2 * CIN)   v = W2[(tap * CIN + ci) * CIN + (co - CIN)];
    }
    wl[idx] = f2bf(v);
}

// ---------------------------------------------------------------------------
// Kernel 3: fused threshold + dual 3x3 conv (implicit GEMM, bf16 WMMA)
//           + active mask + spatial sum. One WG per (b, t, 8-row tile).
// ---------------------------------------------------------------------------
__global__ void __launch_bounds__(256)
spnet_conv_kernel(const float* __restrict__ x,
                  const unsigned short* __restrict__ wl,
                  float* __restrict__ pooledSum) {
    extern __shared__ unsigned char smem[];
    unsigned short* xs_s  = (unsigned short*)smem;            // [10][58][24] bf16
    unsigned short* wl_s  = xs_s + XS_ELEMS;                  // [9][48][32] bf16
    float*          act_s = (float*)(wl_s + WL_ELEMS);        // [448]
    float*          colsum = act_s + SITES_PER_WG;            // [48]

    const int tid = threadIdx.x;
    const int b   = blockIdx.z;
    const int t   = blockIdx.y;
    const int r0  = blockIdx.x * ROWS_PER_TILE;

    // ---- Stage 1: thresholded bf16 activation slab (with halo + ch pad) ----
    const float* xb = x + (size_t)b * CIN * TT * HH * WW;
    for (int i = tid; i < XS_ELEMS; i += 256) {
        int hh  = i / (LCOLS * LCH);
        int rem = i % (LCOLS * LCH);
        int ww  = rem / LCH;
        int c   = rem % LCH;
        int gh  = r0 - 1 + hh;
        int gw  = ww - 1;
        float v = 0.0f;
        if (c < CIN && gh >= 0 && gh < HH && gw >= 0 && gw < WW)
            v = xb[(((size_t)c * TT + t) * HH + gh) * WW + gw];
        if (v < THRESH) v = 0.0f;                 // sparsify in fp32, exact
        xs_s[i] = f2bf(v);
    }
    // ---- Stage weights into LDS (as dwords) ----
    for (int i = tid; i < WL_ELEMS / 2; i += 256)
        ((unsigned int*)wl_s)[i] = ((const unsigned int*)wl)[i];
    if (tid < NPAD) colsum[tid] = 0.0f;
    __syncthreads();

    // ---- Stage 2: per-site active flags (any of 17 center channels != 0) ----
    for (int s = tid; s < SITES_PER_WG; s += 256) {
        int hh = s / WW + 1, ww = s % WW + 1;
        const unsigned int* pp = (const unsigned int*)(xs_s + (hh * LCOLS + ww) * LCH);
        unsigned int o = 0;
        #pragma unroll
        for (int k = 0; k < 9; ++k) o |= pp[k];   // covers ci 0..17 (17 is pad=0)
        act_s[s] = o ? 1.0f : 0.0f;
    }
    __syncthreads();

    // ---- Stage 3: implicit GEMM over 16-site M-tiles ----
    const int  lane = tid & 31;
    const int  wave = tid >> 5;
    const int  nlo  = lane & 15;
    const bool hi   = lane >= 16;
    const u16x8 z8  = {0, 0, 0, 0, 0, 0, 0, 0};

    float c0 = 0.0f, c1 = 0.0f, c2 = 0.0f;       // per-lane column partials

    for (int m = wave; m < MTILES; m += 8) {
        int site = m * 16 + nlo;                  // A-matrix row M for this lane
        int hr   = site / WW;
        int h0   = hr + 1;
        int w0   = site - hr * WW + 1;

        v8f a0 = {}, a1 = {}, a2 = {};
        #pragma unroll
        for (int tap = 0; tap < 9; ++tap) {
            const int dh = tap / 3 - 1, dw = tap % 3 - 1;
            const unsigned short* p = xs_s + ((h0 + dh) * LCOLS + (w0 + dw)) * LCH;
            Frag16 A;
            if (!hi) { A.h[0] = *(const u16x8*)(p);     A.h[1] = *(const u16x8*)(p + 16); }
            else     { A.h[0] = *(const u16x8*)(p + 8); A.h[1] = z8; }  // K 24..31 pad

            a0 = __builtin_amdgcn_wmma_f32_16x16x32_bf16(
                     false, A.v, false, load_b_frag(wl_s, tap, nlo, hi),
                     (short)0, a0, false, false);
            a1 = __builtin_amdgcn_wmma_f32_16x16x32_bf16(
                     false, A.v, false, load_b_frag(wl_s, tap, 16 + nlo, hi),
                     (short)0, a1, false, false);
            a2 = __builtin_amdgcn_wmma_f32_16x16x32_bf16(
                     false, A.v, false, load_b_frag(wl_s, tap, 32 + nlo, hi),
                     (short)0, a2, false, false);
        }

        // D layout: VGPR r holds row M=r (lanes<16) / M=r+8 (lanes>=16), col=nlo.
        // Mask rule: global col n < 17  ->  W1 output, multiply by active(site).
        int mbase = m * 16 + (hi ? 8 : 0);
        #pragma unroll
        for (int r = 0; r < 8; ++r) {
            float a = act_s[mbase + r];
            c0 += a * a0[r];                          // n = nlo      (<17: masked)
            c1 += ((nlo == 0) ? a : 1.0f) * a1[r];    // n = 16+nlo   (only n=16 masked)
            c2 += a2[r];                              // n = 32+nlo   (unmasked / junk-0)
        }
    }

    atomicAdd(&colsum[nlo],      c0);
    atomicAdd(&colsum[16 + nlo], c1);
    atomicAdd(&colsum[32 + nlo], c2);
    __syncthreads();

    // y1[co] lives in col co, y2[co] in col 17+co.
    if (tid < CIN)
        atomicAdd(&pooledSum[b * CIN + tid], colsum[tid] + colsum[CIN + tid]);
}

// ---------------------------------------------------------------------------
// Kernel 4: mean + classifier   out[b][j] = (pooled/N) . Wfc[j] + bfc[j]
// ---------------------------------------------------------------------------
__global__ void spnet_fc_kernel(const float* __restrict__ pooledSum,
                                const float* __restrict__ Wfc,
                                const float* __restrict__ bfc,
                                float* __restrict__ out) {
    int idx = blockIdx.x * blockDim.x + threadIdx.x;
    if (idx >= BATCH * 60) return;
    int b = idx / 60, j = idx % 60;
    const float invN = 1.0f / (float)(TT * HH * WW);
    float s = bfc[j];
    #pragma unroll
    for (int co = 0; co < CIN; ++co)
        s += pooledSum[b * CIN + co] * invN * Wfc[j * CIN + co];
    out[idx] = s;
}

// ---------------------------------------------------------------------------
extern "C" void kernel_launch(void* const* d_in, const int* in_sizes, int n_in,
                              void* d_out, int out_size, void* d_ws, size_t ws_size,
                              hipStream_t stream) {
    const float* x   = (const float*)d_in[0];
    const float* W1  = (const float*)d_in[1];
    const float* W2  = (const float*)d_in[2];
    const float* Wfc = (const float*)d_in[3];
    const float* bfc = (const float*)d_in[4];
    float* out = (float*)d_out;

    float*          pooledSum = (float*)d_ws;                              // 544 f32
    unsigned short* wl        = (unsigned short*)((char*)d_ws + 4096);     // 27648 B

    spnet_zero_kernel<<<dim3(1), dim3(256), 0, stream>>>(pooledSum);
    spnet_wprep_kernel<<<dim3((9 * NPAD * KPAD + 255) / 256), dim3(256), 0, stream>>>(
        W1, W2, wl);

    dim3 grid(HH / ROWS_PER_TILE, TT, BATCH);   // (7, 48, 32)
    spnet_conv_kernel<<<grid, dim3(256), SMEM_BYTES, stream>>>(x, wl, pooledSum);

    spnet_fc_kernel<<<dim3((BATCH * 60 + 255) / 256), dim3(256), 0, stream>>>(
        pooledSum, Wfc, bfc, out);
}
